// MambaSSM_59803124629719
// MI455X (gfx1250) — matrix operand order repaired and checked by reading, not compiled, not run on hardware
//
#include <hip/hip_runtime.h>

// ---------------------------------------------------------------------------
// Problem constants (MambaSSM: B=8, T=2048, D_MODEL=1024, D_SSM=1024)
// ---------------------------------------------------------------------------
#define BB 8
#define TT 2048
#define DM 1024
#define DS 1024
#define MM (BB * TT)          // 16384 rows
#define KK 1024               // shared K for all GEMMs
#define NCH 32                // scan chunks
#define CHL (TT / NCH)        // 64 steps per chunk
#define BK 64                 // K-slab per LDS stage (2 WMMA K-steps)

typedef __bf16 bf16_t;
typedef __attribute__((ext_vector_type(16))) __bf16 v16bf;
typedef __attribute__((ext_vector_type(4)))  __bf16 v4bf;
typedef __attribute__((ext_vector_type(8)))  float  v8f;
typedef __attribute__((ext_vector_type(4)))  unsigned su4;
typedef __attribute__((ext_vector_type(8)))  unsigned su8;

union FragU { v16bf v; uint4 q[2]; };

// ---------------------------------------------------------------------------
// 1) fp32 -> bf16 (vectorized by 4)
// ---------------------------------------------------------------------------
__global__ __launch_bounds__(256) void cvt_f32_bf16(const float* __restrict__ in,
                                                    bf16_t* __restrict__ out, int n4) {
  int i = blockIdx.x * 256 + threadIdx.x;
  if (i >= n4) return;
  float4 v = ((const float4*)in)[i];
  v4bf o;
  o.x = (__bf16)v.x; o.y = (__bf16)v.y; o.z = (__bf16)v.z; o.w = (__bf16)v.w;
  ((v4bf*)out)[i] = o;
}

// ---------------------------------------------------------------------------
// 2) LDS-tiled transpose + convert: W (K x Nfull, f32) -> Wt (Nsel x K, bf16)
//    grid = (Nsel/32, K/32), block = 256 (mapped 32x8)
// ---------------------------------------------------------------------------
__global__ __launch_bounds__(256) void transpose_cvt(const float* __restrict__ W,
                                                     bf16_t* __restrict__ Wt,
                                                     int K, int Nfull) {
  __shared__ float t[32][33];
  const int tx = threadIdx.x & 31;
  const int ty = threadIdx.x >> 5;        // 0..7
  const int kb = blockIdx.y * 32;
  const int nb = blockIdx.x * 32;
#pragma unroll
  for (int i = 0; i < 32; i += 8)
    t[ty + i][tx] = W[(size_t)(kb + ty + i) * Nfull + nb + tx];
  __syncthreads();
#pragma unroll
  for (int i = 0; i < 32; i += 8)
    Wt[(size_t)(nb + ty + i) * K + kb + tx] = (__bf16)t[tx][ty + i];
}

// ---------------------------------------------------------------------------
// TDM helper: DMA a 128-row x BK-col bf16 tile (row stride K elems) into LDS.
// D# group0: count=1 | lds_addr | 57-bit global addr | type=2 ("image").
// Issued by one wave only; tracked on TENSORcnt.
// ---------------------------------------------------------------------------
__device__ __forceinline__ void tdm_load_tile(unsigned lds_off,
                                              const bf16_t* gptr,
                                              su8 g1) {
  const unsigned long long ga = (unsigned long long)gptr;
  su4 g0;
  g0.x = 1u;                                                  // count = 1
  g0.y = lds_off;                                             // LDS byte address
  g0.z = (unsigned)ga;                                        // global addr [31:0]
  g0.w = ((unsigned)(ga >> 32) & 0x01FFFFFFu) | 0x80000000u;  // [56:32] | type=2
  asm volatile("tensor_load_to_lds %0, %1" :: "s"(g0), "s"(g1) : "memory");
}

// group1 (uniform across the kernel): workgroup_mask=0, data_size=2B,
// huge tensor dims (tiles are always interior -> OOB never fires),
// tile = BK x 128, tensor_dim0_stride = K (elements).
__device__ __forceinline__ su8 tdm_group1(unsigned K) {
  su8 g1;
  g1.s0 = 0x00010000u;              // data_size = 1 (2 bytes)
  g1.s1 = 0x00000000u;              // tensor_dim0[15:0] << 16 (dim0 = 0x40000000)
  g1.s2 = 0x00004000u;              // tensor_dim0[31:16]; tensor_dim1[15:0]<<16 = 0
  g1.s3 = ((unsigned)BK << 16) | 0x4000u; // tile_dim0 = BK; tensor_dim1[31:16]
  g1.s4 = 128u;                     // tile_dim1 = 128 rows; tile_dim2 = 0
  g1.s5 = K;                        // tensor_dim0_stride [31:0]
  g1.s6 = 0u;                       // stride hi, dim1_stride lo
  g1.s7 = 0u;                       // dim1_stride hi
  return g1;
}

// ---------------------------------------------------------------------------
// 3) bf16 WMMA GEMM: C(M x N f32, ldc) = A(M x K bf16) * Bt(N x K bf16)^T + bias
//    128x128 block tile, BK=64, 8 waves (2 M x 4 N), 64x32 per wave.
//    A/B tiles staged via TENSOR_LOAD_TO_LDS, double-buffered.
// ---------------------------------------------------------------------------
__global__ __launch_bounds__(256) void gemm_bf16_wmma(const bf16_t* __restrict__ A,
                                                      const bf16_t* __restrict__ Bt,
                                                      const float* __restrict__ bias,
                                                      float* __restrict__ C,
                                                      int K, int ldc) {
  __shared__ bf16_t As[2][128][BK];   // 2 x 16 KB
  __shared__ bf16_t Bs[2][128][BK];   // 2 x 16 KB

  const int tid  = threadIdx.x;
  const int lane = tid & 31;
  const int wave = tid >> 5;
  const int wm   = wave >> 2;      // 0..1  (M dir, 64 rows each)
  const int wn   = wave & 3;       // 0..3  (N dir, 32 cols each)
  const int half = lane >> 4;      // 0/1
  const int r    = lane & 15;
  const int bm   = blockIdx.y * 128;
  const int bn   = blockIdx.x * 128;

  // Scalar (wave-uniform) issuer flag so non-issuing waves *branch around*
  // the TDM op (TDM ignores EXEC, so an EXEC-mask guard would not work).
  const bool issuer = (__builtin_amdgcn_readfirstlane(tid) < 32);
  const su8 g1 = tdm_group1((unsigned)K);

  const unsigned ldsA0 = (unsigned)(size_t)&As[0][0][0];
  const unsigned ldsA1 = (unsigned)(size_t)&As[1][0][0];
  const unsigned ldsB0 = (unsigned)(size_t)&Bs[0][0][0];
  const unsigned ldsB1 = (unsigned)(size_t)&Bs[1][0][0];

  v8f acc[4][2] = {};

  const int iters = K / BK;
  if (issuer) {   // prologue: stage slab 0 into buffer 0
    tdm_load_tile(ldsA0, A  + (size_t)bm * K, g1);
    tdm_load_tile(ldsB0, Bt + (size_t)bn * K, g1);
  }

  for (int it = 0; it < iters; ++it) {
    const int buf = it & 1;
    if (issuer) {
      if (it + 1 < iters) {   // stage slab it+1 into the other buffer
        const size_t knext = (size_t)(it + 1) * BK;
        const unsigned la = (buf ? ldsA0 : ldsA1);
        const unsigned lb = (buf ? ldsB0 : ldsB1);
        tdm_load_tile(la, A  + (size_t)bm * K + knext, g1);
        tdm_load_tile(lb, Bt + (size_t)bn * K + knext, g1);
        __builtin_amdgcn_s_wait_tensorcnt(2);  // slab `it` done; it+1 in flight
      } else {
        __builtin_amdgcn_s_wait_tensorcnt(0);
      }
    }
    __syncthreads();   // buffer `buf` now valid for all waves

#pragma unroll
    for (int kk = 0; kk < BK; kk += 32) {
      // B fragments: lane half 0 -> K 0..15, half 1 -> K 16..31 (contiguous)
      FragU b0, b1;
      {
        const int c0 = wn * 32 + r;
        b0.q[0] = *(const uint4*)&Bs[buf][c0][kk + half * 16];
        b0.q[1] = *(const uint4*)&Bs[buf][c0][kk + half * 16 + 8];
        const int c1 = c0 + 16;
        b1.q[0] = *(const uint4*)&Bs[buf][c1][kk + half * 16];
        b1.q[1] = *(const uint4*)&Bs[buf][c1][kk + half * 16 + 8];
      }
#pragma unroll
      for (int mi = 0; mi < 4; ++mi) {
        // A fragment: half 0 -> K 0..15, half 1 -> K 16..31 (contiguous)
        FragU a;
        const int ar = wm * 64 + mi * 16 + r;
        a.q[0] = *(const uint4*)&As[buf][ar][kk + half * 16];
        a.q[1] = *(const uint4*)&As[buf][ar][kk + half * 16 + 8];
        acc[mi][0] = __builtin_amdgcn_wmma_f32_16x16x32_bf16(
            false, a.v, false, b0.v, (short)0, acc[mi][0], false, false);
        acc[mi][1] = __builtin_amdgcn_wmma_f32_16x16x32_bf16(
            false, a.v, false, b1.v, (short)0, acc[mi][1], false, false);
      }
    }
    __syncthreads();   // all waves done reading `buf` before it is re-filled
  }

  // epilogue: D layout = VGPR v -> row (half*8 + v), col = lane&15
#pragma unroll
  for (int ni = 0; ni < 2; ++ni) {
    const int col = bn + wn * 32 + ni * 16 + r;
    const float bv = bias[col];
#pragma unroll
    for (int mi = 0; mi < 4; ++mi) {
      const int rbase = bm + wm * 64 + mi * 16 + half * 8;
#pragma unroll
      for (int v = 0; v < 8; ++v)
        C[(size_t)(rbase + v) * ldc + col] = acc[mi][ni][v] + bv;
    }
  }
}

// ---------------------------------------------------------------------------
// Scan: s_t = a_t * s_{t-1} + b_t   (linear recurrence)
//   a_t = 1 - g*(1 - tanh(k)),  b_t = g*u,  g = sigmoid(w)
// ---------------------------------------------------------------------------
__device__ __forceinline__ void scan_coeffs(float w, float k, float u,
                                            float& a, float& b) {
  const float g = 1.0f / (1.0f + __expf(-w));
  const float e = __expf(-2.0f * k);
  const float th = (1.0f - e) / (1.0f + e);
  a = 1.0f - g * (1.0f - th);
  b = g * u;
}

// Pass 1: per-chunk composition (A,B). id -> (chunk, batch, channel)
__global__ __launch_bounds__(256) void scan_pass1(const float* __restrict__ wk,
                                                  const float* __restrict__ u,
                                                  float* __restrict__ carryA,
                                                  float* __restrict__ carryB) {
  const int id = blockIdx.x * 256 + threadIdx.x;  // 0 .. NCH*BB*DS-1
  const int d = id & (DS - 1);
  const int b = (id >> 10) & (BB - 1);
  const int c = id >> 13;
  const size_t base = (size_t)b * TT + (size_t)c * CHL;
  float A = 1.0f, Bc = 0.0f;
  for (int t = 0; t < CHL; ++t) {
    const size_t row = base + t;
    float a, bv;
    scan_coeffs(wk[row * 2048 + d], wk[row * 2048 + 1024 + d], u[row * 1024 + d], a, bv);
    A = a * A;
    Bc = a * Bc + bv;
  }
  const int ci = c * (BB * DS) + b * DS + d;
  carryA[ci] = A;
  carryB[ci] = Bc;
}

// Pass 1.5: serial prefix over the NCH chunk carries; carryA[c] <- state before chunk c
__global__ __launch_bounds__(256) void scan_prefix(float* __restrict__ carryA,
                                                   float* __restrict__ carryB) {
  const int id = blockIdx.x * 256 + threadIdx.x;  // 0 .. BB*DS-1
  float s = 0.0f;
  for (int c = 0; c < NCH; ++c) {
    const int ci = c * (BB * DS) + id;
    const float A = carryA[ci];
    const float Bc = carryB[ci];
    carryA[ci] = s;          // start state for chunk c
    s = A * s + Bc;
  }
}

// Pass 2: replay each chunk from its start state, emit ys directly as bf16.
__global__ __launch_bounds__(256) void scan_pass2(const float* __restrict__ wk,
                                                  const float* __restrict__ u,
                                                  const float* __restrict__ carryA,
                                                  bf16_t* __restrict__ ysb) {
  const int id = blockIdx.x * 256 + threadIdx.x;
  const int d = id & (DS - 1);
  const int b = (id >> 10) & (BB - 1);
  const int c = id >> 13;
  const size_t base = (size_t)b * TT + (size_t)c * CHL;
  float s = carryA[c * (BB * DS) + b * DS + d];
  for (int t = 0; t < CHL; ++t) {
    const size_t row = base + t;
    float a, bv;
    scan_coeffs(wk[row * 2048 + d], wk[row * 2048 + 1024 + d], u[row * 1024 + d], a, bv);
    s = a * s + bv;
    ysb[row * 1024 + d] = (__bf16)s;
  }
}

// ---------------------------------------------------------------------------
// Workspace layout (bytes)
// ---------------------------------------------------------------------------
#define OFF_XB    ((size_t)0)                          // 16384*1024 bf16 = 32 MB
#define OFF_WPKT  (OFF_XB   + (size_t)MM * KK * 2)     // 2048*1024 bf16  = 4 MB
#define OFF_WUT   (OFF_WPKT + (size_t)2048 * KK * 2)   // 1024*1024 bf16  = 2 MB
#define OFF_WOT   (OFF_WUT  + (size_t)1024 * KK * 2)   // 1024*1024 bf16  = 2 MB
#define OFF_WK    (OFF_WOT  + (size_t)1024 * KK * 2)   // 16384*2048 f32  = 128 MB
#define OFF_U     (OFF_WK   + (size_t)MM * 2048 * 4)   // 16384*1024 f32  = 64 MB
#define OFF_YSB   (OFF_U    + (size_t)MM * 1024 * 4)   // 16384*1024 bf16 = 32 MB
#define OFF_CA    (OFF_YSB  + (size_t)MM * 1024 * 2)   // 32*8192 f32     = 1 MB
#define OFF_CB    (OFF_CA   + (size_t)NCH * BB * DS * 4)

extern "C" void kernel_launch(void* const* d_in, const int* in_sizes, int n_in,
                              void* d_out, int out_size, void* d_ws, size_t ws_size,
                              hipStream_t stream) {
  const float* x  = (const float*)d_in[0];
  const float* Wp = (const float*)d_in[1];
  const float* bp = (const float*)d_in[2];
  const float* Wu = (const float*)d_in[3];
  const float* bu = (const float*)d_in[4];
  const float* Wo = (const float*)d_in[5];
  const float* bo = (const float*)d_in[6];
  float* out = (float*)d_out;

  char* ws = (char*)d_ws;
  bf16_t* xb    = (bf16_t*)(ws + OFF_XB);
  bf16_t* wpkT  = (bf16_t*)(ws + OFF_WPKT);
  bf16_t* wuT   = (bf16_t*)(ws + OFF_WUT);
  bf16_t* woT   = (bf16_t*)(ws + OFF_WOT);
  float*  wk    = (float*) (ws + OFF_WK);
  float*  u     = (float*) (ws + OFF_U);
  bf16_t* ysb   = (bf16_t*)(ws + OFF_YSB);
  float*  cA    = (float*) (ws + OFF_CA);
  float*  cB    = (float*) (ws + OFF_CB);

  // 1) precision conversion / weight transposition (weights then live in L2)
  {
    const int n4 = (MM * DM) / 4;
    cvt_f32_bf16<<<(n4 + 255) / 256, 256, 0, stream>>>(x, xb, n4);
    transpose_cvt<<<dim3(2048 / 32, KK / 32), 256, 0, stream>>>(Wp, wpkT, KK, 3 * DS);
    transpose_cvt<<<dim3(1024 / 32, KK / 32), 256, 0, stream>>>(Wu, wuT, KK, DS);
    transpose_cvt<<<dim3(1024 / 32, KK / 32), 256, 0, stream>>>(Wo, woT, KK, DM);
  }

  // 2) wk = x @ Wp[:, :2048] + bp[:2048]   (skip the dead third of Wp)
  gemm_bf16_wmma<<<dim3(2048 / 128, MM / 128), 256, 0, stream>>>(xb, wpkT, bp, wk, KK, 2048);
  // 3) u = x @ Wu + bu
  gemm_bf16_wmma<<<dim3(1024 / 128, MM / 128), 256, 0, stream>>>(xb, wuT, bu, u, KK, 1024);

  // 4) chunked linear-recurrence scan -> ys (bf16, WMMA A-operand for final GEMM)
  scan_pass1 <<<(NCH * BB * DS) / 256, 256, 0, stream>>>(wk, u, cA, cB);
  scan_prefix<<<(BB * DS) / 256,       256, 0, stream>>>(cA, cB);
  scan_pass2 <<<(NCH * BB * DS) / 256, 256, 0, stream>>>(wk, u, cA, ysb);

  // 5) out = ys @ Wo + bo
  gemm_bf16_wmma<<<dim3(1024 / 128, MM / 128), 256, 0, stream>>>(ysb, woT, bo, out, KK, 1024);
}
